// GAPModel_57638461112953
// MI455X (gfx1250) — matrix operand order repaired and checked by reading, not compiled, not run on hardware
//
#include <hip/hip_runtime.h>
#include <hip/hip_bf16.h>

typedef __attribute__((ext_vector_type(16))) _Float16 v16h;
typedef __attribute__((ext_vector_type(8)))  float    v8f;

// ---------------- WMMA fragment loaders ----------------
// A-fragment (16x32 f16, ISA 7.12.2 layout): lane L holds row M=L&15;
// lanes 0-15: K = 0..7 and 16..23 ; lanes 16-31: K = 8..15 and 24..31.
__device__ __forceinline__ v16h load_afrag(const _Float16* rowbase, int koff, int lane) {
  const int c0 = koff + ((lane >> 4) << 3);          // halfs; byte offset multiple of 16
  union { v16h v; uint4 q[2]; } u;
  u.q[0] = *(const uint4*)(rowbase + c0);            // K = c0 .. c0+7
  u.q[1] = *(const uint4*)(rowbase + c0 + 16);       // K = c0+16 .. c0+23
  return u.v;
}

// B-fragment from pre-swizzled weights: 16 contiguous halfs per lane (32B).
__device__ __forceinline__ v16h load_bfrag(const _Float16* wb, int frag, int lane) {
  const _Float16* p = wb + frag * 512 + lane * 16;
  union { v16h v; uint4 q[2]; } u;
  u.q[0] = *(const uint4*)(p);
  u.q[1] = *(const uint4*)(p + 8);
  return u.v;
}

// ---------------- small prep kernels ----------------
__global__ void k_degree(const int* __restrict__ dst, float* __restrict__ deg, int nE) {
  int e = blockIdx.x * blockDim.x + threadIdx.x;
  if (e < nE) atomicAdd(&deg[dst[e]], 1.0f);
}

__global__ void k_cast_f2h(const float* __restrict__ X, _Float16* __restrict__ XH, int total) {
  int i = blockIdx.x * blockDim.x + threadIdx.x;
  if (i < total) XH[i] = (_Float16)X[i];
}

// scatter-add float features along edges: thread = (edge, 4-float chunk)
__global__ void k_scatter_f(const float* __restrict__ X, const int* __restrict__ src,
                            const int* __restrict__ dst, float* __restrict__ agg, int nE) {
  int idx = blockIdx.x * blockDim.x + threadIdx.x;
  int e = idx >> 4;
  if (e >= nE) return;
  int f = (idx & 15) * 4;
  const float4 x = *(const float4*)(X + src[e] * 64 + f);
  float* a = agg + dst[e] * 64 + f;
  atomicAdd(a + 0, x.x); atomicAdd(a + 1, x.y);
  atomicAdd(a + 2, x.z); atomicAdd(a + 3, x.w);
}

__global__ void k_scatter_h(const _Float16* __restrict__ X, const int* __restrict__ src,
                            const int* __restrict__ dst, float* __restrict__ agg, int nE) {
  int idx = blockIdx.x * blockDim.x + threadIdx.x;
  int e = idx >> 4;
  if (e >= nE) return;
  int f = (idx & 15) * 4;
  union { uint2 u; _Float16 h[4]; } x;
  x.u = *(const uint2*)(X + src[e] * 64 + f);
  float* a = agg + dst[e] * 64 + f;
  atomicAdd(a + 0, (float)x.h[0]); atomicAdd(a + 1, (float)x.h[1]);
  atomicAdd(a + 2, (float)x.h[2]); atomicAdd(a + 3, (float)x.h[3]);
}

// mean: agg * inv_deg, cast to f16
__global__ void k_scale_cast(const float* __restrict__ agg, const float* __restrict__ deg,
                             _Float16* __restrict__ AH, int total) {
  int i = blockIdx.x * blockDim.x + threadIdx.x;
  if (i >= total) return;
  float d = deg[i >> 6];
  float inv = d > 0.0f ? 1.0f / d : 0.0f;
  AH[i] = (_Float16)(agg[i] * inv);
}

// Pre-swizzle weights into per-lane B-fragment order.
// B (32x16 f16): lane L holds col n=L&15; lanes 0-15: K=kt*32+0..7,16..23, lanes 16-31: K=kt*32+8..15,24..31.
__global__ void k_swizzle_w(const float* __restrict__ Wself, const float* __restrict__ Wneigh,
                            _Float16* __restrict__ WB, int ncols, int ntiles, int total) {
  int idx = blockIdx.x * blockDim.x + threadIdx.x;
  if (idx >= total) return;
  int frag   = idx >> 9;
  int within = idx & 511;
  int lane = within >> 4;
  int j    = within & 15;
  int kt = frag / ntiles;
  int nt = frag % ntiles;
  int k = kt * 32 + ((lane >> 4) << 3) + (j & 7) + ((j >> 3) << 4);
  int n = nt * 16 + (lane & 15);
  float w = (k < 64) ? Wself[k * ncols + n] : Wneigh[(k - 64) * ncols + n];
  WB[idx] = (_Float16)w;
}

// ---------------- fused GEMM: relu([X|A] @ [Wself;Wneigh] + b) -> f16 ----------------
// one wave per 16-row tile; K=128 (4 wmma k-steps), N=64 (4 accumulators)
__global__ void k_gemm_relu(const _Float16* __restrict__ XH, const _Float16* __restrict__ AH,
                            const _Float16* __restrict__ WB, const float* __restrict__ bias,
                            _Float16* __restrict__ OUT, int rowTiles) {
  const int lane = threadIdx.x & 31;
  const int rt = blockIdx.x * (blockDim.x >> 5) + (threadIdx.x >> 5);
  if (rt >= rowTiles) return;                       // wave-uniform: EXEC stays all-ones
  const int rowA = rt * 16 + (lane & 15);
  v8f acc[4] = {};
  #pragma unroll
  for (int kt = 0; kt < 4; ++kt) {
    const _Float16* s = (kt < 2) ? XH : AH;         // K 0-63 = self path, 64-127 = neigh path
    const v16h a = load_afrag(s + rowA * 64, (kt & 1) * 32, lane);
    #pragma unroll
    for (int nt = 0; nt < 4; ++nt) {
      const v16h b = load_bfrag(WB, kt * 4 + nt, lane);
      acc[nt] = __builtin_amdgcn_wmma_f32_16x16x32_f16(false, a, false, b,
                                                       (short)0, acc[nt], false, false);
    }
  }
  const int n0  = lane & 15;
  const int mhi = (lane >> 4) << 3;
  float bn[4];
  #pragma unroll
  for (int nt = 0; nt < 4; ++nt) bn[nt] = bias[nt * 16 + n0];
  #pragma unroll
  for (int v = 0; v < 8; ++v) {
    const int row = rt * 16 + mhi + v;
    #pragma unroll
    for (int nt = 0; nt < 4; ++nt) {
      float x = acc[nt][v] + bn[nt];
      OUT[row * 64 + nt * 16 + n0] = (_Float16)fmaxf(x, 0.0f);
    }
  }
}

// ---------------- FC (64->16) + softmax, fused in-register ----------------
__global__ void k_fc_softmax(const _Float16* __restrict__ H, const _Float16* __restrict__ WB,
                             const float* __restrict__ bias, float* __restrict__ out, int rowTiles) {
  const int lane = threadIdx.x & 31;
  const int rt = blockIdx.x * (blockDim.x >> 5) + (threadIdx.x >> 5);
  if (rt >= rowTiles) return;
  const int rowA = rt * 16 + (lane & 15);
  v8f acc = {};
  #pragma unroll
  for (int kt = 0; kt < 2; ++kt) {
    const v16h a = load_afrag(H + rowA * 64, kt * 32, lane);
    const v16h b = load_bfrag(WB, kt, lane);
    acc = __builtin_amdgcn_wmma_f32_16x16x32_f16(false, a, false, b, (short)0, acc, false, false);
  }
  const int n   = lane & 15;                         // D layout: lane holds column n
  const int mhi = (lane >> 4) << 3;
  const float bb = bias[n];
  #pragma unroll
  for (int v = 0; v < 8; ++v) {
    float x = acc[v] + bb;                           // row = rt*16 + mhi + v
    float mx = x;
    #pragma unroll
    for (int m = 1; m < 16; m <<= 1) mx = fmaxf(mx, __shfl_xor(mx, m, 16));
    float e = __expf(x - mx);
    float s = e;
    #pragma unroll
    for (int m = 1; m < 16; m <<= 1) s += __shfl_xor(s, m, 16);
    out[(rt * 16 + mhi + v) * 16 + n] = e / s;
  }
}

// ---------------- host orchestration ----------------
extern "C" void kernel_launch(void* const* d_in, const int* in_sizes, int n_in,
                              void* d_out, int out_size, void* d_ws, size_t ws_size,
                              hipStream_t stream) {
  (void)n_in; (void)out_size; (void)ws_size;
  const float* features = (const float*)d_in[0];
  const int*   src      = (const int*)d_in[1];
  const int*   dst      = (const int*)d_in[2];
  const float* W_self1  = (const float*)d_in[3];
  const float* W_neigh1 = (const float*)d_in[4];
  const float* b1       = (const float*)d_in[5];
  const float* W_self2  = (const float*)d_in[6];
  const float* W_neigh2 = (const float*)d_in[7];
  const float* b2       = (const float*)d_in[8];
  const float* W_fc     = (const float*)d_in[9];
  const float* b_fc     = (const float*)d_in[10];

  const int N        = in_sizes[0] / 64;   // 100000 (divisible by 16)
  const int nE       = in_sizes[1];        // 1.2M
  const int total    = N * 64;
  const int rowTiles = N / 16;

  char* ws = (char*)d_ws;
  size_t off = 0;
  auto wsalloc = [&](size_t bytes) -> void* {
    void* p = ws + off;
    off = (off + bytes + 255) & ~(size_t)255;
    return p;
  };
  float*    deg  = (float*)   wsalloc((size_t)N * 4);
  float*    agg  = (float*)   wsalloc((size_t)total * 4);
  _Float16* XH   = (_Float16*)wsalloc((size_t)total * 2);
  _Float16* AH   = (_Float16*)wsalloc((size_t)total * 2);
  _Float16* H1   = (_Float16*)wsalloc((size_t)total * 2);
  _Float16* H2   = (_Float16*)wsalloc((size_t)total * 2);
  _Float16* WB1  = (_Float16*)wsalloc(16 * 512 * 2);
  _Float16* WB2  = (_Float16*)wsalloc(16 * 512 * 2);
  _Float16* WBfc = (_Float16*)wsalloc(2 * 512 * 2);

  hipMemsetAsync(deg, 0, (size_t)N * 4, stream);
  hipMemsetAsync(agg, 0, (size_t)total * 4, stream);

  k_swizzle_w<<<(16 * 512 + 255) / 256, 256, 0, stream>>>(W_self1, W_neigh1, WB1, 64, 4, 16 * 512);
  k_swizzle_w<<<(16 * 512 + 255) / 256, 256, 0, stream>>>(W_self2, W_neigh2, WB2, 64, 4, 16 * 512);
  k_swizzle_w<<<(2 * 512 + 255) / 256, 256, 0, stream>>>(W_fc, W_fc, WBfc, 16, 1, 2 * 512);

  k_cast_f2h<<<(total + 255) / 256, 256, 0, stream>>>(features, XH, total);
  k_degree<<<(nE + 255) / 256, 256, 0, stream>>>(dst, deg, nE);

  // layer 1
  k_scatter_f<<<(nE * 16 + 255) / 256, 256, 0, stream>>>(features, src, dst, agg, nE);
  k_scale_cast<<<(total + 255) / 256, 256, 0, stream>>>(agg, deg, AH, total);
  k_gemm_relu<<<(rowTiles + 7) / 8, 256, 0, stream>>>(XH, AH, WB1, b1, H1, rowTiles);

  // layer 2
  hipMemsetAsync(agg, 0, (size_t)total * 4, stream);
  k_scatter_h<<<(nE * 16 + 255) / 256, 256, 0, stream>>>(H1, src, dst, agg, nE);
  k_scale_cast<<<(total + 255) / 256, 256, 0, stream>>>(agg, deg, AH, total);
  k_gemm_relu<<<(rowTiles + 7) / 8, 256, 0, stream>>>(H1, AH, WB2, b2, H2, rowTiles);

  // FC + softmax
  k_fc_softmax<<<(rowTiles + 7) / 8, 256, 0, stream>>>(H2, WBfc, b_fc, (float*)d_out, rowTiles);
}
